// HarmonicLayer_25486335935065
// MI455X (gfx1250) — matrix operand order repaired and checked by reading, not compiled, not run on hardware
//
#include <hip/hip_runtime.h>
#include <hip/hip_bf16.h>
#include <math.h>

// ---------------- problem constants ----------------
#define IN_F    1024
#define OUT_F   50257
#define BATCH_N 2048
#define OUT_PAD 50432          // 197 * 256 : padded N so every 256-col block is full
#define NWB     788            // OUT_PAD / 64 : per-row wave-column-blocks for softmax stats
#define HEXP    10.0f          // min(sqrt(1024), 10)
#define EPS_D   1e-8f

typedef __bf16 bf16;
typedef __attribute__((ext_vector_type(16))) __bf16 v16bf;
typedef __attribute__((ext_vector_type(8)))  __bf16 v8bf;
typedef __attribute__((ext_vector_type(8)))  float  v8f;

union AFrag { v16bf v; v8bf h[2]; };

// ---------------- pass 1: fp32 -> bf16 copy + row squared norms ----------------
// One wave per row; rows >= nvalid (W padding) are written as zeros.
__global__ __launch_bounds__(256) void prep_kernel(const float* __restrict__ src,
                                                   bf16* __restrict__ dst,
                                                   float* __restrict__ sq,
                                                   int nvalid) {
  const int lane = threadIdx.x & 31;
  const int wave = threadIdx.x >> 5;
  const int row  = blockIdx.x * 8 + wave;
  const bool valid = row < nvalid;
  const float* s = src + (size_t)row * IN_F;
  bf16* d = dst + (size_t)row * IN_F;
  float acc = 0.f;
  #pragma unroll 4
  for (int i = lane; i < IN_F; i += 32) {
    float v = valid ? s[i] : 0.f;
    acc += v * v;
    d[i] = (bf16)v;
  }
  #pragma unroll
  for (int off = 1; off < 32; off <<= 1) acc += __shfl_xor(acc, off, 32);
  if (lane == 0) sq[row] = acc;
}

// ---------------- pass 2: WMMA GEMM fused with dist->logit + partial softmax stats ----
// Block = 256 threads = 8 waves, arranged 2 (rows) x 4 (cols).
// Wave tile = 16 rows x 64 cols (4 x wmma_f32_16x16x32_bf16 n-tiles).
__global__ __launch_bounds__(256, 2)
void harmonic_gemm_kernel(const bf16*  __restrict__ xb,    // [BATCH_N][IN_F]
                          const bf16*  __restrict__ wb,    // [OUT_PAD][IN_F] zero-padded
                          const float* __restrict__ xsq,   // [BATCH_N]
                          const float* __restrict__ wsq,   // [OUT_PAD]
                          float*       __restrict__ out,   // [BATCH_N][OUT_F] (logits)
                          float*       __restrict__ stats) // [BATCH_N][NWB][2] = (max,sum)
{
  const int tid  = threadIdx.x;
  const int lane = tid & 31;
  const int wave = tid >> 5;
  const int rg   = wave >> 2;                 // 0..1
  const int cg   = wave & 3;                  // 0..3
  const int mbase = blockIdx.y * 32 + rg * 16;
  const int nbase = blockIdx.x * 256 + cg * 64;
  const int half  = lane >> 4;                // which 16-lane half
  const int lcol  = lane & 15;

  // A: lane holds row (mbase+lcol); halves: K = k0+16*half .. +15 (contiguous)
  const bf16* aRow  = xb + (size_t)(mbase + lcol) * IN_F;
  // B: lane holds column (nbase+16t+lcol); halves: K = k0+16*half .. +15 (contiguous)
  const bf16* bRow0 = wb + (size_t)(nbase + lcol) * IN_F;

  const v8f zero = {0.f, 0.f, 0.f, 0.f, 0.f, 0.f, 0.f, 0.f};
  v8f acc[4] = {zero, zero, zero, zero};

  #pragma unroll 2
  for (int k0 = 0; k0 < IN_F; k0 += 32) {
    AFrag a;
    a.h[0] = *(const v8bf*)(aRow + k0 + 16 * half);
    a.h[1] = *(const v8bf*)(aRow + k0 + 16 * half + 8);
    #pragma unroll
    for (int t = 0; t < 4; ++t) {
      v16bf bfrag = *(const v16bf*)(bRow0 + (size_t)t * 16 * IN_F + k0 + 16 * half);
      acc[t] = __builtin_amdgcn_wmma_f32_16x16x32_bf16(
          /*neg_a=*/false, a.v, /*neg_b=*/false, bfrag,
          /*c_mod=*/(short)0, acc[t], /*reuse_a=*/false, /*reuse_b=*/false);
    }
  }

  // ---- fused epilogue: s = -H*log(max(xsq + wsq - 2*cross, eps)) ----
  // C layout: VGPR r -> row (mbase + 8*half + r), lane lcol -> col (nbase + 16t + lcol)
  float xs[8];
  #pragma unroll
  for (int r = 0; r < 8; ++r) xs[r] = xsq[mbase + 8 * half + r];

  float rmax[8];
  #pragma unroll
  for (int r = 0; r < 8; ++r) rmax[r] = -INFINITY;

  float sv[4][8];
  #pragma unroll
  for (int t = 0; t < 4; ++t) {
    const int o = nbase + 16 * t + lcol;
    const float wsv = wsq[o];
    const bool valid = (o < OUT_F);
    #pragma unroll
    for (int r = 0; r < 8; ++r) {
      float dist = xs[r] + wsv - 2.0f * acc[t][r];
      dist = fmaxf(dist, EPS_D);
      float s = valid ? (-HEXP * logf(dist)) : -INFINITY;
      if (valid) out[(size_t)(mbase + 8 * half + r) * OUT_F + o] = s;
      sv[t][r] = s;
      rmax[r] = fmaxf(rmax[r], s);
    }
  }

  // per-row max across the 16 lanes of each half (rows live in one half only)
  #pragma unroll
  for (int r = 0; r < 8; ++r) {
    #pragma unroll
    for (int off = 1; off < 16; off <<= 1)
      rmax[r] = fmaxf(rmax[r], __shfl_xor(rmax[r], off, 32));
  }

  float rsum[8];
  #pragma unroll
  for (int r = 0; r < 8; ++r) {
    float l = 0.f;
    #pragma unroll
    for (int t = 0; t < 4; ++t) {
      float s = sv[t][r];
      l += (s == -INFINITY) ? 0.f : expf(s - rmax[r]);  // guard padding (and all-pad tiles)
    }
    rsum[r] = l;
  }
  #pragma unroll
  for (int r = 0; r < 8; ++r) {
    #pragma unroll
    for (int off = 1; off < 16; off <<= 1)
      rsum[r] += __shfl_xor(rsum[r], off, 32);
  }

  if (lcol == 0) {
    const int wbk = blockIdx.x * 4 + cg;
    #pragma unroll
    for (int r = 0; r < 8; ++r) {
      size_t idx = ((size_t)(mbase + 8 * half + r) * NWB + wbk) * 2;
      stats[idx]     = rmax[r];
      stats[idx + 1] = rsum[r];
    }
  }
}

// ---------------- pass 3: per-row logsumexp from partial (max,sum) stats ----------------
__global__ __launch_bounds__(256) void lse_kernel(const float* __restrict__ stats,
                                                  float* __restrict__ lse) {
  const int row = blockIdx.x;
  const int tid = threadIdx.x;
  __shared__ float sm[256];

  float m = -INFINITY;
  for (int i = tid; i < NWB; i += 256)
    m = fmaxf(m, stats[((size_t)row * NWB + i) * 2]);
  sm[tid] = m; __syncthreads();
  for (int s = 128; s > 0; s >>= 1) {
    if (tid < s) sm[tid] = fmaxf(sm[tid], sm[tid + s]);
    __syncthreads();
  }
  const float M = sm[0];
  __syncthreads();

  float l = 0.f;
  for (int i = tid; i < NWB; i += 256) {
    float mi = stats[((size_t)row * NWB + i) * 2];
    float li = stats[((size_t)row * NWB + i) * 2 + 1];
    if (mi > -INFINITY) l += expf(mi - M) * li;
  }
  sm[tid] = l; __syncthreads();
  for (int s = 128; s > 0; s >>= 1) {
    if (tid < s) sm[tid] += sm[tid + s];
    __syncthreads();
  }
  if (tid == 0) lse[row] = M + logf(sm[0]);
}

// ---------------- pass 4: out = logits - lse[row] ----------------
__global__ __launch_bounds__(256) void sub_kernel(float* __restrict__ out,
                                                  const float* __restrict__ lse) {
  const int row = blockIdx.y;
  const int col = blockIdx.x * 256 + threadIdx.x;
  if (col < OUT_F) out[(size_t)row * OUT_F + col] -= lse[row];
}

extern "C" void kernel_launch(void* const* d_in, const int* in_sizes, int n_in,
                              void* d_out, int out_size, void* d_ws, size_t ws_size,
                              hipStream_t stream) {
  const float* x = (const float*)d_in[0];   // [2048, 1024]
  const float* w = (const float*)d_in[1];   // [50257, 1024]
  float* out = (float*)d_out;               // [2048, 50257]

  char* ws = (char*)d_ws;
  size_t off = 0;
  bf16* wb    = (bf16*)(ws + off);  off += (size_t)OUT_PAD * IN_F * sizeof(bf16);   // 103.3 MB
  bf16* xb    = (bf16*)(ws + off);  off += (size_t)BATCH_N * IN_F * sizeof(bf16);   //   4.2 MB
  float* wsq  = (float*)(ws + off); off += (size_t)OUT_PAD * sizeof(float);
  float* xsq  = (float*)(ws + off); off += (size_t)BATCH_N * sizeof(float);
  float* stats= (float*)(ws + off); off += (size_t)BATCH_N * NWB * 2 * sizeof(float); // 12.9 MB
  float* lse  = (float*)(ws + off); off += (size_t)BATCH_N * sizeof(float);
  (void)ws_size; (void)in_sizes; (void)n_in; (void)out_size;

  prep_kernel<<<OUT_PAD / 8, 256, 0, stream>>>(w, wb, wsq, OUT_F);
  prep_kernel<<<BATCH_N / 8, 256, 0, stream>>>(x, xb, xsq, BATCH_N);

  dim3 grid_gemm(OUT_PAD / 256, BATCH_N / 32);
  harmonic_gemm_kernel<<<grid_gemm, 256, 0, stream>>>(xb, wb, xsq, wsq, out, stats);

  lse_kernel<<<BATCH_N, 256, 0, stream>>>(stats, lse);

  dim3 grid_sub((OUT_F + 255) / 256, BATCH_N);
  sub_kernel<<<grid_sub, 256, 0, stream>>>(out, lse);
}
